// SyntaxesAttention_21723944583666
// MI455X (gfx1250) — matrix operand order repaired and checked by reading, not compiled, not run on hardware
//
#include <hip/hip_runtime.h>

// Sliding-window attention on gfx1250 (MI455X):
//  - all matmuls via v_wmma_f32_16x16x32_bf16
//  - attention K window staged to LDS via TDM (tensor_load_to_lds, TENSORcnt)
//  - attention V window staged to LDS via global_load_async_to_lds_b128 (ASYNCcnt)
// S=2048, HIDDEN=768, 12 heads x 64 head_dim, window 64, scale 1/8.

#define SLEN   2048
#define HID    768
#define NHEADS 12
#define HDIM   64
#define KSEL   64
#define ASCALE 0.125f

typedef unsigned short u16;
typedef __attribute__((ext_vector_type(16))) __bf16    v16bf;
typedef __attribute__((ext_vector_type(8)))  float     v8f;
typedef __attribute__((ext_vector_type(4)))  unsigned  v4u;
typedef __attribute__((ext_vector_type(8)))  unsigned  v8u;

union FragBF {
    v16bf bf;
    u16   u[16];
    uint4 q[2];
};

__device__ __forceinline__ u16 f32_to_bf16(float f) {
    union { float f; unsigned int u; } v; v.f = f;
    unsigned int u = v.u;
    u += 0x7FFFu + ((u >> 16) & 1u);   // round-to-nearest-even
    return (u16)(u >> 16);
}

// low 32 bits of a generic pointer into LDS == LDS byte offset (gfx aperture)
__device__ __forceinline__ unsigned lds_off(const void* p) {
    return (unsigned)(unsigned long long)p;
}

// async global->LDS 16B copy (GV mode), tracked by ASYNCcnt
__device__ __forceinline__ void async_ld_b128(unsigned ldsByteOff, const void* gptr) {
    asm volatile("global_load_async_to_lds_b128 %0, %1, off"
                 :: "v"(ldsByteOff), "v"(gptr) : "memory");
}
__device__ __forceinline__ void wait_async0() {
    asm volatile("s_wait_asynccnt 0x0" ::: "memory");
}
__device__ __forceinline__ void wait_tensor0() {
    asm volatile("s_wait_tensorcnt 0x0" ::: "memory");
}

// TDM: load a [tile_rows x 64] bf16 tile (row stride HID elems) from global to
// LDS (packed row-major). 2-D descriptor => only D# groups 0 and 1 are needed.
// OOB rows (>= tdim1_rows from tile start) return zeros. All operands uniform.
__device__ __forceinline__ void tdm_load_tile(unsigned ldsByteOff, const void* gptr,
                                              unsigned tdim1_rows, unsigned tile_rows) {
    unsigned long long ga = (unsigned long long)gptr;
    const v4u g0 = {
        1u,                                    // count=1, user D#, no gather
        ldsByteOff,                            // lds_addr (bytes)
        (unsigned)ga,                          // global_addr[31:0]
        ((unsigned)(ga >> 32) & 0x01FFFFFFu) | 0x80000000u  // addr[56:32] | type=2
    };
    const v8u g1 = {
        0x00010000u,                           // data_size=1 (2 bytes/elem)
        (unsigned)HDIM << 16,                  // tensor_dim0[15:0]=64 in [31:16]
        (tdim1_rows & 0xFFFFu) << 16,          // tensor_dim1[15:0] in [31:16]
        (unsigned)HDIM << 16,                  // tensor_dim1 hi=0 | tile_dim0=64
        tile_rows & 0xFFFFu,                   // tile_dim1 | tile_dim2=0
        (unsigned)HID,                         // tensor_dim0_stride[31:0]=768
        0u,                                    // dim0_stride hi | dim1_stride lo
        0u                                     // dim1_stride hi
    };
    asm volatile("tensor_load_to_lds %0, %1" :: "s"(g0), "s"(g1) : "memory");
}

// ---------------------------------------------------------------------------
// f32 -> bf16 conversion
// ---------------------------------------------------------------------------
__global__ void cvt_bf16_kernel(const float* __restrict__ src,
                                u16* __restrict__ dst, int n) {
    int i = blockIdx.x * blockDim.x + threadIdx.x;
    if (i < n) dst[i] = f32_to_bf16(src[i]);
}

// ---------------------------------------------------------------------------
// WMMA GEMM, 32x64 tile per wave (2x4 accumulators): out = A @ W^T.
// Fragment layouts per CDNA5 ISA 7.12.2 (wave32):
//   A 16x32 bf16: lane m = lane&15; u[0..7] = k(base..+7), u[8..15] = k+16..,
//                 base = (lane>>4)*8
//   B 32x16 bf16: lane n = lane&15; u[0..7] = k(base..+7), u[8..15] = k+16.. (same as A)
//   C/D f32 16x16: lane n = lane&15; elem r -> m = r + (lane>>4)*8
// ---------------------------------------------------------------------------
__device__ __forceinline__ void wmma_gemm_2x4(const u16* __restrict__ A,
                                              const u16* __restrict__ W,
                                              int r0, int c0, v8f acc[2][4]) {
    const int lane = threadIdx.x & 31;
    const int l15  = lane & 15;
    const int half = lane >> 4;
    const u16* ap0 = A + (size_t)(r0 + l15) * HID + half * 8;
    const u16* ap1 = ap0 + (size_t)16 * HID;
    const u16* bp0 = W + (size_t)(c0 + l15) * HID + half * 8;

    for (int k0 = 0; k0 < HID; k0 += 32) {
        FragBF a0, a1;
        a0.q[0] = *(const uint4*)(ap0 + k0);
        a0.q[1] = *(const uint4*)(ap0 + k0 + 16);
        a1.q[0] = *(const uint4*)(ap1 + k0);
        a1.q[1] = *(const uint4*)(ap1 + k0 + 16);
        __builtin_prefetch(ap0 + k0 + 64, 0, 0);   // global_prefetch_b8
        __builtin_prefetch(ap1 + k0 + 64, 0, 0);
#pragma unroll
        for (int j = 0; j < 4; ++j) {
            FragBF b;
            const u16* bp = bp0 + (size_t)(16 * j) * HID + k0;
            b.q[0] = *(const uint4*)bp;
            b.q[1] = *(const uint4*)(bp + 16);
            acc[0][j] = __builtin_amdgcn_wmma_f32_16x16x32_bf16(
                false, a0.bf, false, b.bf, (short)0, acc[0][j], false, false);
            acc[1][j] = __builtin_amdgcn_wmma_f32_16x16x32_bf16(
                false, a1.bf, false, b.bf, (short)0, acc[1][j], false, false);
        }
    }
}

// Q/K/V projections fused: blockIdx.z selects weight/output.
// grid = (SLEN/128, HID/64, 3), block = 128 (4 waves, 32x64 tile each).
__global__ __launch_bounds__(128)
void qkv_gemm_kernel(const u16* __restrict__ Xb,
                     const u16* __restrict__ Wqb,
                     const u16* __restrict__ Wkb,
                     const u16* __restrict__ Wvb,
                     u16* __restrict__ Qb,
                     u16* __restrict__ Kb,
                     u16* __restrict__ Vb) {
    const u16* W; u16* O;
    if      (blockIdx.z == 0) { W = Wqb; O = Qb; }
    else if (blockIdx.z == 1) { W = Wkb; O = Kb; }
    else                      { W = Wvb; O = Vb; }

    const int wave = threadIdx.x >> 5;
    const int lane = threadIdx.x & 31;
    const int l15  = lane & 15;
    const int half = lane >> 4;
    const int r0 = blockIdx.x * 128 + wave * 32;
    const int c0 = blockIdx.y * 64;

    v8f acc[2][4] = {};
    wmma_gemm_2x4(Xb, W, r0, c0, acc);

#pragma unroll
    for (int i = 0; i < 2; ++i)
#pragma unroll
        for (int j = 0; j < 4; ++j) {
            u16* op = O + (size_t)(r0 + 16 * i + half * 8) * HID + c0 + 16 * j + l15;
#pragma unroll
            for (int r = 0; r < 8; ++r) op[(size_t)r * HID] = f32_to_bf16(acc[i][j][r]);
        }
}

// Output projection: bf16 attn @ Wo^T -> f32 d_out.
__global__ __launch_bounds__(128)
void oproj_gemm_kernel(const u16* __restrict__ Ab,
                       const u16* __restrict__ Wob,
                       float* __restrict__ out) {
    const int wave = threadIdx.x >> 5;
    const int lane = threadIdx.x & 31;
    const int l15  = lane & 15;
    const int half = lane >> 4;
    const int r0 = blockIdx.x * 128 + wave * 32;
    const int c0 = blockIdx.y * 64;

    v8f acc[2][4] = {};
    wmma_gemm_2x4(Ab, Wob, r0, c0, acc);

#pragma unroll
    for (int i = 0; i < 2; ++i)
#pragma unroll
        for (int j = 0; j < 4; ++j) {
            float* op = out + (size_t)(r0 + 16 * i + half * 8) * HID + c0 + 16 * j + l15;
#pragma unroll
            for (int r = 0; r < 8; ++r) op[(size_t)r * HID] = acc[i][j][r];
        }
}

// ---------------------------------------------------------------------------
// Sliding-window attention. One wave per (16-query block, head).
// Candidate keys start at kbase = max(q0-64, 0); 80-key K tile via TDM,
// 96-key V tile via async b128 copies. OOB / padded rows are zero or real
// (finite) data; the softmax mask and P=0 padding neutralize them.
// grid = (SLEN/16, NHEADS), block = 32.
// ---------------------------------------------------------------------------
__global__ __launch_bounds__(32)
void attn_kernel(const u16* __restrict__ Qb,
                 const u16* __restrict__ Kb,
                 const u16* __restrict__ Vb,
                 u16* __restrict__ Ab) {
    __shared__ __align__(16) u16 Klds[80 * 64];   // [key_local][d]
    __shared__ __align__(16) u16 Vlds[96 * 64];   // [key_local][d]
    __shared__ __align__(16) u16 Plds[16 * 96];   // [query][key_local]

    const int lane = threadIdx.x & 31;
    const int l15  = lane & 15;
    const int half = lane >> 4;
    const int q0    = blockIdx.x * 16;
    const int head  = blockIdx.y;
    const int kbase = (q0 - 64) < 0 ? 0 : (q0 - 64);   // window tile base key

    // ---- K window via TDM: one 80x64 bf16 tile, rows beyond SLEN -> 0 ----
    tdm_load_tile(lds_off(&Klds[0]),
                  Kb + (size_t)kbase * HID + head * HDIM,
                  (unsigned)(SLEN - kbase), 80u);

    // ---- V window via async copies: 96 rows x 128B = 768 chunks (24/lane) ----
    {
        const unsigned vbase = lds_off(&Vlds[0]);
        for (int c = lane; c < 768; c += 32) {
            int row = c >> 3, part = c & 7;
            int key = kbase + row;
            int kc  = key > SLEN - 1 ? SLEN - 1 : key;   // clamp keeps data finite
            async_ld_b128(vbase + (unsigned)c * 16,
                          Vb + (size_t)kc * HID + head * HDIM + part * 8);
        }
    }

    // Q A-fragments (global) for the two 32-wide head_dim chunks
    FragBF qa[2];
#pragma unroll
    for (int dc = 0; dc < 2; ++dc) {
        const u16* p = Qb + (size_t)(q0 + l15) * HID + head * HDIM + dc * 32 + half * 8;
        qa[dc].q[0] = *(const uint4*)p;
        qa[dc].q[1] = *(const uint4*)(p + 16);
    }

    wait_tensor0();   // K tile resident
    wait_async0();    // V tile resident

    // scores: 5 tiles of 16x16, contraction over head_dim (2 WMMAs each)
    v8f sc[5];
#pragma unroll
    for (int t = 0; t < 5; ++t) {
        v8f a = {};
#pragma unroll
        for (int dc = 0; dc < 2; ++dc) {
            FragBF kb;
            const u16* p = Klds + (t * 16 + l15) * HDIM + dc * 32 + half * 8;
            kb.q[0] = *(const uint4*)p;
            kb.q[1] = *(const uint4*)(p + 16);
            a = __builtin_amdgcn_wmma_f32_16x16x32_bf16(
                false, qa[dc].bf, false, kb.bf, (short)0, a, false, false);
        }
        sc[t] = a;
    }

    // mask + scale; row max
    float mx[8], sum[8];
#pragma unroll
    for (int r = 0; r < 8; ++r) { mx[r] = -3.0e38f; sum[r] = 0.0f; }
#pragma unroll
    for (int t = 0; t < 5; ++t) {
#pragma unroll
        for (int r = 0; r < 8; ++r) {
            int m   = r + half * 8;
            int q   = q0 + m;
            int key = kbase + t * 16 + l15;
            bool valid = (key <= q) && (key >= q - (KSEL - 1));
            float s = valid ? sc[t][r] * ASCALE : -3.0e38f;
            sc[t][r] = s;
            mx[r] = fmaxf(mx[r], s);
        }
    }
    // each score row lives in one 16-lane half -> xor 1,2,4,8 stays in-half
#pragma unroll
    for (int r = 0; r < 8; ++r)
#pragma unroll
        for (int d = 1; d < 16; d <<= 1)
            mx[r] = fmaxf(mx[r], __shfl_xor(mx[r], d, 32));

#pragma unroll
    for (int t = 0; t < 5; ++t)
#pragma unroll
        for (int r = 0; r < 8; ++r) {
            float p = __expf(sc[t][r] - mx[r]);
            sc[t][r] = p;
            sum[r] += p;
        }
#pragma unroll
    for (int r = 0; r < 8; ++r)
#pragma unroll
        for (int d = 1; d < 16; d <<= 1)
            sum[r] += __shfl_xor(sum[r], d, 32);

    float inv[8];
#pragma unroll
    for (int r = 0; r < 8; ++r) inv[r] = 1.0f / sum[r];

    // write normalized P (bf16) to LDS [16][96]; pad keys 80..95 = 0
#pragma unroll
    for (int t = 0; t < 5; ++t)
#pragma unroll
        for (int r = 0; r < 8; ++r)
            Plds[(r + half * 8) * 96 + t * 16 + l15] = f32_to_bf16(sc[t][r] * inv[r]);
#pragma unroll
    for (int r = 0; r < 8; ++r)
        Plds[(r + half * 8) * 96 + 80 + l15] = 0;
    __syncthreads();

    // out = P(16x96) * V(96x64): 3 K-chunks x 4 N-tiles, V from LDS
    v8f o[4] = {};
#pragma unroll
    for (int t = 0; t < 3; ++t) {
        FragBF pa;
        const u16* lp = Plds + l15 * 96 + 32 * t + half * 8;
        pa.q[0] = *(const uint4*)lp;
        pa.q[1] = *(const uint4*)(lp + 16);
#pragma unroll
        for (int nt = 0; nt < 4; ++nt) {
            FragBF vb;
#pragma unroll
            for (int e = 0; e < 16; ++e)
                vb.u[e] = Vlds[(32 * t + half * 8 + (e & 7) + (e >> 3) * 16) * HDIM + nt * 16 + l15];
            o[nt] = __builtin_amdgcn_wmma_f32_16x16x32_bf16(
                false, pa.bf, false, vb.bf, (short)0, o[nt], false, false);
        }
    }

    u16* op = Ab + (size_t)(q0 + half * 8) * HID + head * HDIM + l15;
#pragma unroll
    for (int nt = 0; nt < 4; ++nt)
#pragma unroll
        for (int r = 0; r < 8; ++r)
            op[(size_t)r * HID + nt * 16] = f32_to_bf16(o[nt][r]);
}

// ---------------------------------------------------------------------------
extern "C" void kernel_launch(void* const* d_in, const int* in_sizes, int n_in,
                              void* d_out, int out_size, void* d_ws, size_t ws_size,
                              hipStream_t stream) {
    const float* X  = (const float*)d_in[0];
    const float* Wq = (const float*)d_in[1];
    const float* Wk = (const float*)d_in[2];
    const float* Wv = (const float*)d_in[3];
    const float* Wo = (const float*)d_in[4];
    float* out = (float*)d_out;

    const size_t SX = (size_t)SLEN * HID;  // 1,572,864 elems
    const size_t SW = (size_t)HID * HID;   //   589,824 elems

    // workspace layout (bf16), total ~20.4 MB; all 256B-aligned sections
    char* ws = (char*)d_ws;
    u16* Xb  = (u16*)ws;
    u16* Wqb = (u16*)(ws + 2 * SX);
    u16* Wkb = Wqb + SW;
    u16* Wvb = Wkb + SW;
    u16* Wob = Wvb + SW;
    u16* Qb  = Wob + SW;
    u16* Kb  = Qb + SX;
    u16* Vb  = Kb + SX;
    u16* Ab  = Vb + SX;

    cvt_bf16_kernel<<<(int)((SX + 255) / 256), 256, 0, stream>>>(X,  Xb,  (int)SX);
    cvt_bf16_kernel<<<(int)((SW + 255) / 256), 256, 0, stream>>>(Wq, Wqb, (int)SW);
    cvt_bf16_kernel<<<(int)((SW + 255) / 256), 256, 0, stream>>>(Wk, Wkb, (int)SW);
    cvt_bf16_kernel<<<(int)((SW + 255) / 256), 256, 0, stream>>>(Wv, Wvb, (int)SW);
    cvt_bf16_kernel<<<(int)((SW + 255) / 256), 256, 0, stream>>>(Wo, Wob, (int)SW);

    qkv_gemm_kernel<<<dim3(SLEN / 128, HID / 64, 3), 128, 0, stream>>>(
        Xb, Wqb, Wkb, Wvb, Qb, Kb, Vb);

    attn_kernel<<<dim3(SLEN / 16, NHEADS), 32, 0, stream>>>(Qb, Kb, Vb, Ab);

    oproj_gemm_kernel<<<dim3(SLEN / 128, HID / 64), 128, 0, stream>>>(
        Ab, Wob, out);
}
